// LVPGLayer_41661182771241
// MI455X (gfx1250) — compile-verified, matrix operand order checked by reading, not run on hardware
//
#include <hip/hip_runtime.h>

// ---------------- types ----------------
typedef __bf16 bf16_t;
typedef bf16_t v16bf __attribute__((ext_vector_type(16)));
typedef float  v8f   __attribute__((ext_vector_type(8)));

// Problem constants (from the reference)
#define TT   196
#define TPAD 208      // 13 tiles of 16
#define BB   64
#define CC   1024
#define EE   512
#define NN   (CC * EE)   // 524288

// ---------------------------------------------------------------------------
// Kernel 1: emb = LN(embed_table[tok]); h = relu(emb @ W1^T + b1)  -> bf16[B][E]
// One block per batch element. Tiny (0.03 GFLOP) — plain VALU fp32.
// ---------------------------------------------------------------------------
__global__ void __launch_bounds__(256)
k_embed_h(const float* __restrict__ etab, const int* __restrict__ toks,
          const float* __restrict__ g, const float* __restrict__ bln,
          const float* __restrict__ W1, const float* __restrict__ b1,
          bf16_t* __restrict__ h_out)
{
    const int b   = blockIdx.x;
    const int tid = threadIdx.x;
    __shared__ float embn[EE];
    __shared__ float red[256];

    const int tok = toks[b];
    const float* row = etab + (size_t)tok * EE;
    float x0 = row[tid], x1 = row[tid + 256];

    red[tid] = x0 + x1;
    __syncthreads();
    for (int off = 128; off > 0; off >>= 1) {
        if (tid < off) red[tid] += red[tid + off];
        __syncthreads();
    }
    const float mu = red[0] * (1.0f / EE);
    __syncthreads();

    const float d0 = x0 - mu, d1 = x1 - mu;
    red[tid] = d0 * d0 + d1 * d1;
    __syncthreads();
    for (int off = 128; off > 0; off >>= 1) {
        if (tid < off) red[tid] += red[tid + off];
        __syncthreads();
    }
    const float rs = rsqrtf(red[0] * (1.0f / EE) + 1e-5f);
    __syncthreads();

    embn[tid]       = d0 * rs * g[tid]       + bln[tid];
    embn[tid + 256] = d1 * rs * g[tid + 256] + bln[tid + 256];
    __syncthreads();

#pragma unroll
    for (int jj = 0; jj < 2; ++jj) {
        const int j = tid + jj * 256;
        const float* wr = W1 + (size_t)j * EE;
        float acc = b1[j];
#pragma unroll 4
        for (int k = 0; k < EE; ++k) acc = fmaf(embn[k], wr[k], acc);
        h_out[(size_t)b * EE + j] = (bf16_t)fmaxf(acc, 0.0f);
    }
}

// ---------------------------------------------------------------------------
// Kernel 2: featn = LN(feat) over C, stored bf16 as [b][t][c] (t padded to 208)
// ---------------------------------------------------------------------------
__global__ void __launch_bounds__(256)
k_featn(const float* __restrict__ feat, const float* __restrict__ g,
        const float* __restrict__ bln, bf16_t* __restrict__ fo)
{
    const int t = blockIdx.x, b = blockIdx.y, tid = threadIdx.x;
    __shared__ float red[256];
    const float* x = feat + ((size_t)t * BB + b) * CC;

    float v[4];
    float s = 0.f;
#pragma unroll
    for (int i = 0; i < 4; ++i) { v[i] = x[tid + 256 * i]; s += v[i]; }
    red[tid] = s;
    __syncthreads();
    for (int off = 128; off > 0; off >>= 1) {
        if (tid < off) red[tid] += red[tid + off];
        __syncthreads();
    }
    const float mu = red[0] * (1.0f / CC);
    __syncthreads();
    s = 0.f;
#pragma unroll
    for (int i = 0; i < 4; ++i) { v[i] -= mu; s += v[i] * v[i]; }
    red[tid] = s;
    __syncthreads();
    for (int off = 128; off > 0; off >>= 1) {
        if (tid < off) red[tid] += red[tid + off];
        __syncthreads();
    }
    const float rs = rsqrtf(red[0] * (1.0f / CC) + 1e-5f);

    bf16_t* o = fo + ((size_t)b * TPAD + t) * CC;
#pragma unroll
    for (int i = 0; i < 4; ++i) {
        const int c = tid + 256 * i;
        o[c] = (bf16_t)(v[i] * rs * g[c] + bln[c]);
    }
}

// ---------------------------------------------------------------------------
// Kernel 3: w = relu(h @ W2^T + b2)   M=64(B), K=512, N=524288
// bf16 WMMA, fp32 accumulate. h (64KB bf16) staged in LDS once per block.
// 8 waves/block, each wave owns 16 N columns and all 4 M-tiles (4 accumulators).
// HBM-bound on the W2 stream (1.07 GB) by design.
// ---------------------------------------------------------------------------
__global__ void __launch_bounds__(256)
k_gemm_w(const bf16_t* __restrict__ h, const float* __restrict__ W2,
         const float* __restrict__ b2, bf16_t* __restrict__ w_out)
{
    __shared__ __align__(16) bf16_t hs[BB * EE];   // 64 KB
    const int tid = threadIdx.x;

    // cooperative stage of h: 4096 uint4, 16 per thread
#pragma unroll
    for (int i = 0; i < 16; ++i)
        ((uint4*)hs)[i * 256 + tid] = ((const uint4*)h)[i * 256 + tid];
    __syncthreads();

    const int wid  = tid >> 5;
    const int lane = tid & 31;
    const int nlo  = lane & 15;
    const int kb   = (lane >> 4) * 8;            // ISA 16-bit A/B lane K-base
    const int n    = blockIdx.x * 128 + wid * 16 + nlo;

    const float* w2row = W2 + (size_t)n * EE;

    v8f acc0 = {}, acc1 = {}, acc2 = {}, acc3 = {};

    for (int kc = 0; kc < EE; kc += 32) {
        // prefetch next K-chunk of this W2 row into cache
        __builtin_prefetch(w2row + kc + 128, 0, 1);

        // B-matrix tile: W2 fp32 -> bf16, lane holds column n, K striped
        const float4 f0 = *(const float4*)(w2row + kc + kb);
        const float4 f1 = *(const float4*)(w2row + kc + kb + 4);
        const float4 f2 = *(const float4*)(w2row + kc + 16 + kb);
        const float4 f3 = *(const float4*)(w2row + kc + 16 + kb + 4);
        v16bf bm;
        bm[0]=(bf16_t)f0.x;  bm[1]=(bf16_t)f0.y;  bm[2]=(bf16_t)f0.z;  bm[3]=(bf16_t)f0.w;
        bm[4]=(bf16_t)f1.x;  bm[5]=(bf16_t)f1.y;  bm[6]=(bf16_t)f1.z;  bm[7]=(bf16_t)f1.w;
        bm[8]=(bf16_t)f2.x;  bm[9]=(bf16_t)f2.y;  bm[10]=(bf16_t)f2.z; bm[11]=(bf16_t)f2.w;
        bm[12]=(bf16_t)f3.x; bm[13]=(bf16_t)f3.y; bm[14]=(bf16_t)f3.z; bm[15]=(bf16_t)f3.w;

        // A-matrix tiles from LDS (h rows), 4 M-tiles of 16
        union { uint4 u[2]; v16bf v; } ca;
#define LOAD_A(mt, amv)                                                        \
        ca.u[0] = *(const uint4*)(hs + ((mt * 16 + nlo) * EE + kc + kb));      \
        ca.u[1] = *(const uint4*)(hs + ((mt * 16 + nlo) * EE + kc + 16 + kb)); \
        const v16bf amv = ca.v;

        { LOAD_A(0, a0)
          acc0 = __builtin_amdgcn_wmma_f32_16x16x32_bf16(false, a0, false, bm,
                                                         (short)0, acc0, false, false); }
        { LOAD_A(1, a1)
          acc1 = __builtin_amdgcn_wmma_f32_16x16x32_bf16(false, a1, false, bm,
                                                         (short)0, acc1, false, false); }
        { LOAD_A(2, a2)
          acc2 = __builtin_amdgcn_wmma_f32_16x16x32_bf16(false, a2, false, bm,
                                                         (short)0, acc2, false, false); }
        { LOAD_A(3, a3)
          acc3 = __builtin_amdgcn_wmma_f32_16x16x32_bf16(false, a3, false, bm,
                                                         (short)0, acc3, false, false); }
#undef LOAD_A
    }

    // Epilogue: +bias, relu, bf16, scatter to w[b][n] (n = e*C + c, natural order)
    const float bias = b2[n];
    const int mhi = (lane >> 4) * 8;    // D layout: VGPR r -> M = r (+8 for hi lanes)
#pragma unroll
    for (int r = 0; r < 8; ++r) {
        const int m0 = r + mhi;
        w_out[(size_t)(m0)      * NN + n] = (bf16_t)fmaxf(acc0[r] + bias, 0.f);
        w_out[(size_t)(m0 + 16) * NN + n] = (bf16_t)fmaxf(acc1[r] + bias, 0.f);
        w_out[(size_t)(m0 + 32) * NN + n] = (bf16_t)fmaxf(acc2[r] + bias, 0.f);
        w_out[(size_t)(m0 + 48) * NN + n] = (bf16_t)fmaxf(acc3[r] + bias, 0.f);
    }
}

// ---------------------------------------------------------------------------
// Kernel 4: out[t,b,e] = sum_c featn[b][t][c] * w[b][e*C + c]
// 64 batched GEMMs [196x1024]@[1024x512]^T via bf16 WMMA.
// All 8 waves share one A tile (16 t-rows x 1024 c = 32 KB bf16): staged once
// per block into LDS with GLOBAL_LOAD_ASYNC_TO_LDS_B128 (ASYNCcnt path),
// then fed to WMMA via ds_load_b128. B (wmap, 1MB/b) stays L2-resident.
// ---------------------------------------------------------------------------
__global__ void __launch_bounds__(256)
k_out(const bf16_t* __restrict__ featn, const bf16_t* __restrict__ wmap,
      float* __restrict__ out)
{
    __shared__ __align__(16) bf16_t at[16 * CC];   // 32 KB A tile
    const int tid  = threadIdx.x;
    const int wid  = tid >> 5;
    const int lane = tid & 31;
    const int nlo  = lane & 15;
    const int kb   = (lane >> 4) * 8;

    const int b  = blockIdx.z;
    const int t0 = blockIdx.y * 16;
    const int e  = blockIdx.x * 128 + wid * 16 + nlo;

    // ---- async stage of the shared A tile: 2048 x b128, 8 per thread ----
    {
        const char* src = (const char*)(featn + ((size_t)b * TPAD + t0) * CC);
        // Low 32 bits of the generic pointer are the LDS byte address
        // (shared-aperture flat addresses carry the LDS offset in addr[31:0]).
        const unsigned ldsbase = (unsigned)(unsigned long long)(const void*)at;
#pragma unroll
        for (int i = 0; i < 8; ++i) {
            const unsigned off = (unsigned)(i * 256 + tid) * 16u;
            asm volatile("global_load_async_to_lds_b128 %0, %1, off"
                         :: "v"(ldsbase + off), "v"(src + off)
                         : "memory");
        }
        asm volatile("s_wait_asynccnt 0x0" ::: "memory");
    }
    __syncthreads();

    const bf16_t* arow = at + (size_t)nlo * CC;                    // A: row t, K=c (LDS)
    const bf16_t* brow = wmap + (size_t)b * NN + (size_t)e * CC;   // B: col e, K=c

    v8f acc = {};
    for (int kc = 0; kc < CC; kc += 32) {
        union { uint4 u[2]; v16bf v; } ca, cb;
        ca.u[0] = *(const uint4*)(arow + kc + kb);          // ds_load_b128
        ca.u[1] = *(const uint4*)(arow + kc + 16 + kb);
        cb.u[0] = *(const uint4*)(brow + kc + kb);          // global_load_b128
        cb.u[1] = *(const uint4*)(brow + kc + 16 + kb);
        acc = __builtin_amdgcn_wmma_f32_16x16x32_bf16(false, ca.v, false, cb.v,
                                                      (short)0, acc, false, false);
    }

    const int mhi = (lane >> 4) * 8;
#pragma unroll
    for (int r = 0; r < 8; ++r) {
        const int t = t0 + r + mhi;
        if (t < TT) out[((size_t)t * BB + b) * EE + e] = acc[r];
    }
}

// ---------------------------------------------------------------------------
// Launch: ws layout = h bf16 (64KB) | featn bf16 (26MB) | wmap bf16 (64MB)
// ---------------------------------------------------------------------------
extern "C" void kernel_launch(void* const* d_in, const int* in_sizes, int n_in,
                              void* d_out, int out_size, void* d_ws, size_t ws_size,
                              hipStream_t stream)
{
    const float* feat = (const float*)d_in[0];
    const int*   toks = (const int*)  d_in[1];
    const float* etab = (const float*)d_in[2];
    const float* eg   = (const float*)d_in[3];
    const float* eb   = (const float*)d_in[4];
    const float* fg   = (const float*)d_in[5];
    const float* fb   = (const float*)d_in[6];
    const float* W1   = (const float*)d_in[7];
    const float* b1   = (const float*)d_in[8];
    const float* W2   = (const float*)d_in[9];
    const float* b2   = (const float*)d_in[10];
    float* out = (float*)d_out;

    char* ws = (char*)d_ws;
    bf16_t* h     = (bf16_t*)ws;                                   // 65,536 B
    bf16_t* fn    = (bf16_t*)(ws + (size_t)65536);                 // 27,262,976 B
    bf16_t* wmapb = (bf16_t*)(ws + (size_t)65536 + 27262976u);     // 67,108,864 B

    k_embed_h<<<BB, 256, 0, stream>>>(etab, toks, eg, eb, W1, b1, h);
    k_featn<<<dim3(TT, BB), 256, 0, stream>>>(feat, fg, fb, fn);
    k_gemm_w<<<NN / 128, 256, 0, stream>>>(h, W2, b2, wmapb);
    k_out<<<dim3(EE / 128, TPAD / 16, BB), 256, 0, stream>>>(fn, wmapb, out);
}
